// Bridge_2d_23106924052678
// MI455X (gfx1250) — matrix operand order repaired and checked by reading, not compiled, not run on hardware
//
#include <hip/hip_runtime.h>
#include <cstdint>
#include <cstddef>

typedef _Float16 h16;
typedef __attribute__((ext_vector_type(16))) _Float16 v16h;
typedef __attribute__((ext_vector_type(8)))  _Float16 v8h;
typedef __attribute__((ext_vector_type(8)))  float    v8f;

#define NSAMP   262144
#define NSTEP   8
#define NBLK    8192          /* NSAMP / 32 samples per block (2 waves x 16) */

/* ---------------- workspace byte offsets ---------------- */
static constexpr size_t B_PL  = 0;                 /* float[NBLK]  per-block log sums  */
static constexpr size_t B_PP  = 32768;             /* float[NBLK]  per-block pen sums  */
static constexpr size_t B_VEC = 65536;             /* float[4][4][64] b,gw,gb,bw      */
static constexpr size_t B_W0C = B_VEC + 4096;      /* float[2][64] columns of W0      */
static constexpr size_t B_W0P = B_W0C + 512;       /* h16[64][32]  W0 padded (K->32)  */
static constexpr size_t B_W1  = B_W0P + 4096;      /* h16[64][64]                     */
static constexpr size_t B_W2  = B_W1  + 8192;      /* h16[64][64]                     */
static constexpr size_t B_W3P = B_W2  + 8192;      /* h16[16][64]  W3 padded (N->16)  */

/* ---------------- branch-free transcendentals on v_exp/v_rcp ---------------- */
__device__ __forceinline__ float fast_rcp(float x) {
#if __has_builtin(__builtin_amdgcn_rcpf)
  return __builtin_amdgcn_rcpf(x);
#else
  return 1.0f / x;
#endif
}
__device__ __forceinline__ float fast_exp2(float x) {
#if __has_builtin(__builtin_amdgcn_exp2f)
  return __builtin_amdgcn_exp2f(x);
#else
  return exp2f(x);
#endif
}
__device__ __forceinline__ float fast_tanh(float x) {
#if __has_builtin(__builtin_amdgcn_tanhf)
  return __builtin_amdgcn_tanhf(x);
#else
  /* tanh(x) = 1 - 2/(e^{2x}+1); saturates to +-1 without branches */
  float e = fast_exp2(2.8853900817779268f * x);   /* 2*log2(e) */
  return 1.0f - 2.0f * fast_rcp(e + 1.0f);
#endif
}
__device__ __forceinline__ float sigm(float x) {
  return fast_rcp(1.0f + fast_exp2(-1.4426950408889634f * x));
}

/* ---------------- small vector helpers ---------------- */
__device__ __forceinline__ v8f fzero() {
  v8f v;
#pragma unroll
  for (int i = 0; i < 8; ++i) v[i] = 0.f;
  return v;
}
__device__ __forceinline__ v8f vsplat(float x) {
  v8f v;
#pragma unroll
  for (int i = 0; i < 8; ++i) v[i] = x;
  return v;
}
__device__ __forceinline__ v16h hzero() {
  v16h v;
#pragma unroll
  for (int i = 0; i < 16; ++i) v[i] = (h16)0.f;
  return v;
}
__device__ __forceinline__ v8f vtanh(v8f u) {
  v8f r;
#pragma unroll
  for (int i = 0; i < 8; ++i) r[i] = fast_tanh(u[i]);
  return r;
}

__device__ __forceinline__ v8f wmma(v16h a, v16h b, v8f c) {
  return __builtin_amdgcn_wmma_f32_16x16x32_f16(false, a, false, b, (short)0, c, false, false);
}

/* 16 halves from two 16B LDS loads: p0[0..7], then p1[0..7] */
__device__ __forceinline__ v16h ld16(const h16* p0, const h16* p1) {
  union { v16h v; v8h h[2]; } u;
  u.h[0] = *(const v8h*)p0;
  u.h[1] = *(const v8h*)p1;
  return u.v;
}
/* B operand (K=32 x N=16) from weights stored row-major by output-neuron n (rowStride k's).
   lane n = lane&15 (+16*nt global), lane<16: k runs {0..7,16..23}; lane>=16: k runs {8..15,24..31}. */
__device__ __forceinline__ v16h loadB(const h16* base, int rowStride, int lane, int nt, int kc) {
  int n  = (lane & 15) + nt * 16;
  int kh = lane >> 4;
  const h16* p = base + n * rowStride + kc * 32 + kh * 8;
  return ld16(p, p + 16);
}
/* A operand (M=16 x K=32) from per-wave activation tile act[m][64] (row-major).
   lane<16: m=lane, k runs {0..7,16..23}; lane>=16: m=lane-16, k runs {8..15,24..31}. */
__device__ __forceinline__ v16h loadA(const h16* act, int lane, int kc) {
  int m  = lane & 15;
  int kh = lane >> 4;
  const h16* p = act + m * 64 + kc * 32 + kh * 8;
  return ld16(p, p + 16);
}
/* Scatter a WMMA C tile (f32) back into act[m][64] as f16.
   C layout: n = lane&15 (+16*nt), m = vgpr + 8*(lane>=16). */
__device__ __forceinline__ void storeAct(h16* act, int lane, int nt, v8f v) {
  int n  = nt * 16 + (lane & 15);
  int mb = (lane >> 4) * 8;
#pragma unroll
  for (int r = 0; r < 8; ++r) act[(mb + r) * 64 + n] = (h16)v[r];
}

/* ---------------- shared-memory view handed to the net evaluators ---------------- */
struct Shm {
  const h16*   W0p;   /* [64][32] */
  const h16*   W1;    /* [64][64] */
  const h16*   W2;    /* [64][64] */
  const h16*   W3p;   /* [16][64] */
  const float* vec;   /* [4][4][64] : b,gw,gb,bw per layer */
  const float* w0c0;  /* W0[:,0] */
  const float* w0c1;  /* W0[:,1] */
  h16*         act;   /* this wave: 6 channels x 16x64 f16 */
  float*       outw;  /* this wave: 6 channels x 2 x 16 f32 */
};
#define VEC(L, P, ng) S.vec[(((L) * 4) + (P)) * 64 + (ng)]

/* ---------------- value-only network evaluation (forward ODE) ---------------- */
__device__ __forceinline__ void net_value(const Shm& S, int lane, float x0, float x1, float t,
                                          float& f0, float& f1) {
  int nloc = lane & 15;
  int hi   = lane >> 4;
  /* layer 0: A = [x0 x1 0 ... 0] built in registers (K padded to 32) */
  v16h a0 = hzero();
  a0[0] = (hi == 0) ? (h16)x0 : (h16)0.f;
  a0[1] = (hi == 0) ? (h16)x1 : (h16)0.f;
#pragma unroll
  for (int nt = 0; nt < 4; ++nt) {
    v16h b  = loadB(S.W0p, 32, lane, nt, 0);
    v8f lin = wmma(a0, b, fzero());
    int ng  = nt * 16 + nloc;
    float gate = sigm(VEC(0, 1, ng) * t + VEC(0, 2, ng));
    v8f u = (lin + vsplat(VEC(0, 0, ng))) * vsplat(gate) + vsplat(VEC(0, 3, ng) * t);
    storeAct(S.act, lane, nt, vtanh(u));
  }
  /* layers 1,2 */
  for (int L = 1; L <= 2; ++L) {
    const h16* W = (L == 1) ? S.W1 : S.W2;
    v16h A0 = loadA(S.act, lane, 0);
    v16h A1 = loadA(S.act, lane, 1);
#pragma unroll
    for (int nt = 0; nt < 4; ++nt) {
      v16h b0 = loadB(W, 64, lane, nt, 0);
      v16h b1 = loadB(W, 64, lane, nt, 1);
      v8f lin = wmma(A0, b0, fzero());
      lin     = wmma(A1, b1, lin);
      int ng  = nt * 16 + nloc;
      float gate = sigm(VEC(L, 1, ng) * t + VEC(L, 2, ng));
      v8f u = (lin + vsplat(VEC(L, 0, ng))) * vsplat(gate) + vsplat(VEC(L, 3, ng) * t);
      storeAct(S.act, lane, nt, vtanh(u));
    }
  }
  /* layer 3 (64 -> 2, N padded to 16), then transpose via LDS */
  {
    v16h A0 = loadA(S.act, lane, 0);
    v16h A1 = loadA(S.act, lane, 1);
    v16h b0 = loadB(S.W3p, 64, lane, 0, 0);
    v16h b1 = loadB(S.W3p, 64, lane, 0, 1);
    v8f lin = wmma(A0, b0, fzero());
    lin     = wmma(A1, b1, lin);
    float gate = sigm(VEC(3, 1, nloc) * t + VEC(3, 2, nloc));
    v8f u = (lin + vsplat(VEC(3, 0, nloc))) * vsplat(gate) + vsplat(VEC(3, 3, nloc) * t);
    if (nloc < 2) {
#pragma unroll
      for (int r = 0; r < 8; ++r) S.outw[nloc * 16 + hi * 8 + r] = u[r];
    }
  }
  f0 = S.outw[0 * 16 + nloc];
  f1 = S.outw[1 * 16 + nloc];
}

/* -------- 2nd-order forward-mode jet evaluation (backward ODE RHS) --------
   channels: 0=v, 1=d/dx0, 2=d/dx1, 3=d2/dx0^2, 4=d2/dx0dx1, 5=d2/dx1^2 */
__device__ __forceinline__ void net_jet(const Shm& S, int lane, float x0, float x1,
                                        float sc0, float sc1, float t,
                                        float& f0, float& f1, float& dlog,
                                        float& dsc0, float& dsc1, float& Spen) {
  int nloc = lane & 15;
  int hi   = lane >> 4;
  /* layer 0: value via WMMA; tangents are sample-independent (gate*W0 columns) */
  v16h a0 = hzero();
  a0[0] = (hi == 0) ? (h16)x0 : (h16)0.f;
  a0[1] = (hi == 0) ? (h16)x1 : (h16)0.f;
#pragma unroll
  for (int nt = 0; nt < 4; ++nt) {
    v16h b  = loadB(S.W0p, 32, lane, nt, 0);
    v8f lin = wmma(a0, b, fzero());
    int ng  = nt * 16 + nloc;
    float gate = sigm(VEC(0, 1, ng) * t + VEC(0, 2, ng));
    float g1 = S.w0c0[ng] * gate;
    float g2 = S.w0c1[ng] * gate;
    v8f u = (lin + vsplat(VEC(0, 0, ng))) * vsplat(gate) + vsplat(VEC(0, 3, ng) * t);
    v8f y  = vtanh(u);
    v8f s  = vsplat(1.f) - y * y;
    v8f m2 = vsplat(2.f) * y * s;
    storeAct(S.act + 0 * 1024, lane, nt, y);
    storeAct(S.act + 1 * 1024, lane, nt, s * vsplat(g1));
    storeAct(S.act + 2 * 1024, lane, nt, s * vsplat(g2));
    storeAct(S.act + 3 * 1024, lane, nt, m2 * vsplat(-(g1 * g1)));
    storeAct(S.act + 4 * 1024, lane, nt, m2 * vsplat(-(g1 * g2)));
    storeAct(S.act + 5 * 1024, lane, nt, m2 * vsplat(-(g2 * g2)));
  }
  /* layers 1,2: six WMMA chains per 16-col tile, then tanh-jet algebra */
  for (int L = 1; L <= 2; ++L) {
    const h16* W = (L == 1) ? S.W1 : S.W2;
    v16h A[6][2];
#pragma unroll
    for (int c = 0; c < 6; ++c) {
      A[c][0] = loadA(S.act + c * 1024, lane, 0);
      A[c][1] = loadA(S.act + c * 1024, lane, 1);
    }
#pragma unroll
    for (int nt = 0; nt < 4; ++nt) {
      v16h b0 = loadB(W, 64, lane, nt, 0);
      v16h b1 = loadB(W, 64, lane, nt, 1);
      v8f u[6];
#pragma unroll
      for (int c = 0; c < 6; ++c) {
        v8f acc = wmma(A[c][0], b0, fzero());
        u[c]    = wmma(A[c][1], b1, acc);
      }
      int ng = nt * 16 + nloc;
      float gate = sigm(VEC(L, 1, ng) * t + VEC(L, 2, ng));
      v8f g  = vsplat(gate);
      v8f uv = (u[0] + vsplat(VEC(L, 0, ng))) * g + vsplat(VEC(L, 3, ng) * t);
      v8f u1 = u[1] * g, u2 = u[2] * g, u3 = u[3] * g, u4 = u[4] * g, u5 = u[5] * g;
      v8f y  = vtanh(uv);
      v8f s  = vsplat(1.f) - y * y;
      v8f m2 = vsplat(2.f) * y * s;
      storeAct(S.act + 0 * 1024, lane, nt, y);
      storeAct(S.act + 1 * 1024, lane, nt, s * u1);
      storeAct(S.act + 2 * 1024, lane, nt, s * u2);
      storeAct(S.act + 3 * 1024, lane, nt, s * u3 - m2 * u1 * u1);
      storeAct(S.act + 4 * 1024, lane, nt, s * u4 - m2 * u1 * u2);
      storeAct(S.act + 5 * 1024, lane, nt, s * u5 - m2 * u2 * u2);
    }
  }
  /* layer 3: 6 channels, N padded to 16; transpose per-sample via LDS */
  {
    v16h b0 = loadB(S.W3p, 64, lane, 0, 0);
    v16h b1 = loadB(S.W3p, 64, lane, 0, 1);
    float gate = sigm(VEC(3, 1, nloc) * t + VEC(3, 2, nloc));
#pragma unroll
    for (int c = 0; c < 6; ++c) {
      v16h Ac0 = loadA(S.act + c * 1024, lane, 0);
      v16h Ac1 = loadA(S.act + c * 1024, lane, 1);
      v8f acc = wmma(Ac0, b0, fzero());
      acc     = wmma(Ac1, b1, acc);
      v8f u;
      if (c == 0)
        u = (acc + vsplat(VEC(3, 0, nloc))) * vsplat(gate) + vsplat(VEC(3, 3, nloc) * t);
      else
        u = acc * vsplat(gate);
      if (nloc < 2) {
#pragma unroll
        for (int r = 0; r < 8; ++r) S.outw[c * 32 + nloc * 16 + hi * 8 + r] = u[r];
      }
    }
  }
  /* gather this lane's sample (m = nloc) and assemble the backward RHS */
  float F0   = S.outw[0 * 32 + 0 * 16 + nloc];
  float F1   = S.outw[0 * 32 + 1 * 16 + nloc];
  float J00  = S.outw[1 * 32 + 0 * 16 + nloc];  /* df0/dx0 */
  float J10  = S.outw[1 * 32 + 1 * 16 + nloc];  /* df1/dx0 */
  float J01  = S.outw[2 * 32 + 0 * 16 + nloc];  /* df0/dx1 */
  float J11  = S.outw[2 * 32 + 1 * 16 + nloc];  /* df1/dx1 */
  float H110 = S.outw[3 * 32 + 0 * 16 + nloc];
  float H111 = S.outw[3 * 32 + 1 * 16 + nloc];
  float H120 = S.outw[4 * 32 + 0 * 16 + nloc];
  float H121 = S.outw[4 * 32 + 1 * 16 + nloc];
  float H220 = S.outw[5 * 32 + 0 * 16 + nloc];
  float H221 = S.outw[5 * 32 + 1 * 16 + nloc];
  f0 = F0; f1 = F1;
  dlog = -(J00 + J11);
  float gd0 = H110 + H121;        /* d(div)/dx0 */
  float gd1 = H120 + H221;        /* d(div)/dx1 */
  float jt0 = sc0 * J00 + sc1 * J10;
  float jt1 = sc0 * J01 + sc1 * J11;
  dsc0 = -gd0 - jt0;
  dsc1 = -gd1 - jt1;
  float e0 = F0 + sc0, e1 = F1 + sc1;
  Spen = e0 * e0 + e1 * e1;
}

/* ---------------- weight staging / conversion kernel ---------------- */
__global__ __launch_bounds__(256) void k_prep(
    const float* W0, const float* b0, const float* gw0, const float* gb0, const float* bw0,
    const float* W1, const float* b1, const float* gw1, const float* gb1, const float* bw1,
    const float* W2, const float* b2, const float* gw2, const float* gb2, const float* bw2,
    const float* W3, const float* b3, const float* gw3, const float* gb3, const float* bw3,
    unsigned char* wsb) {
  h16* dW0p = (h16*)(wsb + B_W0P);
  h16* dW1  = (h16*)(wsb + B_W1);
  h16* dW2  = (h16*)(wsb + B_W2);
  h16* dW3p = (h16*)(wsb + B_W3P);
  float* vec = (float*)(wsb + B_VEC);
  float* w0c = (float*)(wsb + B_W0C);
  int tid = threadIdx.x;
  for (int i = tid; i < 64 * 32; i += 256) {
    int n = i >> 5, k = i & 31;
    dW0p[i] = (k < 2) ? (h16)W0[n * 2 + k] : (h16)0.f;
  }
  for (int i = tid; i < 64 * 64; i += 256) { dW1[i] = (h16)W1[i]; dW2[i] = (h16)W2[i]; }
  for (int i = tid; i < 16 * 64; i += 256) {
    int n = i >> 6, k = i & 63;
    dW3p[i] = (n < 2) ? (h16)W3[n * 64 + k] : (h16)0.f;
  }
  for (int i = tid; i < 64; i += 256) { w0c[i] = W0[i * 2 + 0]; w0c[64 + i] = W0[i * 2 + 1]; }
  const float* pb[4]  = { b0, b1, b2, b3 };
  const float* pgw[4] = { gw0, gw1, gw2, gw3 };
  const float* pgb[4] = { gb0, gb1, gb2, gb3 };
  const float* pbw[4] = { bw0, bw1, bw2, bw3 };
  for (int i = tid; i < 4 * 64; i += 256) {
    int L = i >> 6, n = i & 63;
    int len = (L == 3) ? 2 : 64;
    vec[(L * 4 + 0) * 64 + n] = (n < len) ? pb[L][n]  : 0.f;
    vec[(L * 4 + 1) * 64 + n] = (n < len) ? pgw[L][n] : 0.f;
    vec[(L * 4 + 2) * 64 + n] = (n < len) ? pgb[L][n] : 0.f;
    vec[(L * 4 + 3) * 64 + n] = (n < len) ? pbw[L][n] : 0.f;
  }
}

/* ---------------- main ODE kernel: 2 waves/block, 16 samples/wave ---------------- */
__global__ __launch_bounds__(64) void k_ode(const float* __restrict__ x,
                                            const unsigned char* __restrict__ wsb,
                                            float* __restrict__ out,
                                            float* __restrict__ partL,
                                            float* __restrict__ partP) {
  __shared__ h16  sW0p[64 * 32];
  __shared__ h16  sW1[64 * 64];
  __shared__ h16  sW2[64 * 64];
  __shared__ h16  sW3p[16 * 64];
  __shared__ float sVec[4 * 4 * 64];
  __shared__ float sW0c[2 * 64];
  __shared__ h16  sAct[2][6 * 16 * 64];
  __shared__ float sOutS[2][6 * 32];
  __shared__ float sRedL[64], sRedP[64];

  int tid = threadIdx.x;
  /* cooperative staging of weights into LDS */
  {
    const uint32_t* g;
    uint32_t* d;
    g = (const uint32_t*)(wsb + B_W0P); d = (uint32_t*)sW0p;
    for (int i = tid; i < 1024; i += 64) d[i] = g[i];
    g = (const uint32_t*)(wsb + B_W1);  d = (uint32_t*)sW1;
    for (int i = tid; i < 2048; i += 64) d[i] = g[i];
    g = (const uint32_t*)(wsb + B_W2);  d = (uint32_t*)sW2;
    for (int i = tid; i < 2048; i += 64) d[i] = g[i];
    g = (const uint32_t*)(wsb + B_W3P); d = (uint32_t*)sW3p;
    for (int i = tid; i < 512; i += 64) d[i] = g[i];
    g = (const uint32_t*)(wsb + B_VEC); d = (uint32_t*)sVec;
    for (int i = tid; i < 1024; i += 64) d[i] = g[i];
    g = (const uint32_t*)(wsb + B_W0C); d = (uint32_t*)sW0c;
    for (int i = tid; i < 128; i += 64) d[i] = g[i];
  }
  __syncthreads();

  int wv = tid >> 5, lane = tid & 31;
  Shm S { sW0p, sW1, sW2, sW3p, sVec, &sW0c[0], &sW0c[64], sAct[wv], sOutS[wv] };
  int nloc = lane & 15;
  int samp = blockIdx.x * 32 + wv * 16 + nloc;

  float z0 = x[2 * samp + 0];
  float z1 = x[2 * samp + 1];

  const float hf = 1.0f / 8.0f;
  /* forward flow integration */
  for (int k = 0; k < NSTEP; ++k) {
    float t0 = hf * (float)k;
    float a0, a1, b0, b1, c0, c1, d0, d1;
    net_value(S, lane, z0, z1, t0, a0, a1);
    net_value(S, lane, z0 + 0.5f * hf * a0, z1 + 0.5f * hf * a1, t0 + 0.5f * hf, b0, b1);
    net_value(S, lane, z0 + 0.5f * hf * b0, z1 + 0.5f * hf * b1, t0 + 0.5f * hf, c0, c1);
    net_value(S, lane, z0 + hf * c0, z1 + hf * c1, t0 + hf, d0, d1);
    z0 += hf * (a0 + 2.f * b0 + 2.f * c0 + d0) * (1.f / 6.f);
    z1 += hf * (a1 + 2.f * b1 + 2.f * c1 + d1) * (1.f / 6.f);
  }
  /* terminal conditions */
  float lg  = -1.8378770664093453f - 0.5f * (z0 * z0 + z1 * z1);  /* -log(2*pi) - |z|^2/2 */
  float sc0 = -z0, sc1 = -z1;
  float pen = 0.f;

  const float hb = -1.0f / 8.0f;
  float t = 1.0f;
  for (int k = 0; k < NSTEP; ++k) {
    float f0, f1, dl, d0, d1, Sp;
    net_jet(S, lane, z0, z1, sc0, sc1, t, f0, f1, dl, d0, d1, Sp);
    float k1x = hb * f0, k1y = hb * f1, k1l = hb * dl, k1s = hb * d0, k1r = hb * d1, S1 = Sp;
    net_jet(S, lane, z0 + 0.5f * k1x, z1 + 0.5f * k1y, sc0 + 0.5f * k1s, sc1 + 0.5f * k1r,
            t + 0.5f * hb, f0, f1, dl, d0, d1, Sp);
    float k2x = hb * f0, k2y = hb * f1, k2l = hb * dl, k2s = hb * d0, k2r = hb * d1, S2 = Sp;
    net_jet(S, lane, z0 + 0.5f * k2x, z1 + 0.5f * k2y, sc0 + 0.5f * k2s, sc1 + 0.5f * k2r,
            t + 0.5f * hb, f0, f1, dl, d0, d1, Sp);
    float k3x = hb * f0, k3y = hb * f1, k3l = hb * dl, k3s = hb * d0, k3r = hb * d1, S3 = Sp;
    net_jet(S, lane, z0 + k3x, z1 + k3y, sc0 + k3s, sc1 + k3r, t + hb,
            f0, f1, dl, d0, d1, Sp);
    float k4x = hb * f0, k4y = hb * f1, k4l = hb * dl, k4s = hb * d0, k4r = hb * d1, S4 = Sp;
    z0  += (k1x + 2.f * k2x + 2.f * k3x + k4x) * (1.f / 6.f);
    z1  += (k1y + 2.f * k2y + 2.f * k3y + k4y) * (1.f / 6.f);
    lg  += (k1l + 2.f * k2l + 2.f * k3l + k4l) * (1.f / 6.f);
    sc0 += (k1s + 2.f * k2s + 2.f * k3s + k4s) * (1.f / 6.f);
    sc1 += (k1r + 2.f * k2r + 2.f * k3r + k4r) * (1.f / 6.f);
    pen += hb * (S1 + 2.f * S2 + 2.f * S3 + S4) * (1.f / 6.f);
    t += hb;
  }

  if (lane < 16) out[1 + samp] = lg;

  /* deterministic per-block partial sums */
  sRedL[tid] = (lane < 16) ? lg  : 0.f;
  sRedP[tid] = (lane < 16) ? pen : 0.f;
  __syncthreads();
  if (tid == 0) {
    float sL = 0.f, sP = 0.f;
    for (int i = 0; i < 64; ++i) { sL += sRedL[i]; sP += sRedP[i]; }
    partL[blockIdx.x] = sL;
    partP[blockIdx.x] = sP;
  }
}

/* ---------------- final deterministic reduction ---------------- */
__global__ __launch_bounds__(256) void k_fin(const float* __restrict__ partL,
                                             const float* __restrict__ partP,
                                             float* __restrict__ out) {
  __shared__ float rL[256], rP[256];
  int tid = threadIdx.x;
  float sL = 0.f, sP = 0.f;
  for (int i = tid; i < NBLK; i += 256) { sL += partL[i]; sP += partP[i]; }
  rL[tid] = sL; rP[tid] = sP;
  __syncthreads();
  for (int s = 128; s > 0; s >>= 1) {
    if (tid < s) { rL[tid] += rL[tid + s]; rP[tid] += rP[tid + s]; }
    __syncthreads();
  }
  if (tid == 0) {
    float meanL  = rL[0] / (float)NSAMP;
    float meanP  = rP[0] / (float)NSAMP;
    float bridge = -meanP;                /* bridge_pen = -cat_var[0,-1] */
    out[0]         = -meanL * 10.0f + bridge;
    out[1 + NSAMP] = bridge;
  }
}

extern "C" void kernel_launch(void* const* d_in, const int* in_sizes, int n_in,
                              void* d_out, int out_size, void* d_ws, size_t ws_size,
                              hipStream_t stream) {
  (void)in_sizes; (void)n_in; (void)out_size; (void)ws_size;
  const float* x = (const float*)d_in[0];
  const float* P[20];
  for (int i = 0; i < 20; ++i) P[i] = (const float*)d_in[1 + i];
  unsigned char* wsb = (unsigned char*)d_ws;
  float* out = (float*)d_out;

  k_prep<<<1, 256, 0, stream>>>(P[0], P[1], P[2], P[3], P[4],
                                P[5], P[6], P[7], P[8], P[9],
                                P[10], P[11], P[12], P[13], P[14],
                                P[15], P[16], P[17], P[18], P[19], wsb);

  float* partL = (float*)(wsb + B_PL);
  float* partP = (float*)(wsb + B_PP);
  k_ode<<<NBLK, 64, 0, stream>>>(x, wsb, out, partL, partP);
  k_fin<<<1, 256, 0, stream>>>(partL, partP, out);
}
